// Decoder_78812649882416
// MI455X (gfx1250) — matrix operand order repaired and checked by reading, not compiled, not run on hardware
//
#include <hip/hip_runtime.h>
#include <hip/hip_bf16.h>

// ---------------------------------------------------------------------------
// LAS-style attention LSTM decoder for gfx1250 (MI455X).
// Persistent-kernel design: 32 WGs x 256 threads, global barrier between
// phases, bf16 WMMA (v_wmma_f32_16x16x32_bf16) for all GEMMs, fp32 VALU
// attention. Weights pre-swizzled into WMMA per-lane layout in prologue.
// ---------------------------------------------------------------------------

typedef __attribute__((ext_vector_type(16))) __bf16 v16bf;
typedef __attribute__((ext_vector_type(8)))  __bf16 v8bf;
typedef __attribute__((ext_vector_type(8)))  float  v8f;

#define NWG   32
#define NTHR  256
#define BATCH 128
#define SLEN  500
#define TSTEP 249      // T-1
#define HDIM  512
#define KDIM  128
#define VDIM  128
#define VOCAB 1000
#define NEGE  (-1e9f)

// element counts
#define H1N (BATCH*HDIM)     // 65536
#define H2N (BATCH*KDIM)     // 16384
#define W1G ((size_t)32*36*512)  // one-gate stride in w1sw elems (589824)
#define W2G ((size_t)8*20*512)   // one-gate stride in w2sw elems (81920)

struct DecArgs {
    const float* key;      // [S,B,K]
    const float* values;   // [S,B,V]
    const int*   lens;     // [B]
    const int*   text;     // [T,B]
    const float* b_ih1; const float* b_hh1;
    const float* b_ih2; const float* b_hh2;
    const float* b_out;
    const __bf16* w1;      // [128 nt][36 kt][32 lane][16]
    const __bf16* w2;      // [32 nt][20 kt][32 lane][16]
    const __bf16* w3;      // [63 nt][ 8 kt][32 lane][16]
    const __bf16* etab;    // [VOCAB][H] bf16
    __bf16* h1bf;          // 2 x [B][H]
    __bf16* h2bf;          // 2 x [B][K]
    __bf16* ctxbf;         // 2 x [B][V]
    float*  c1;            // [B][H]
    float*  c2;            // [B][K]
    int*    bar;           // [0]=cnt [1]=gen
    float*  out;           // [B][T-1][VOCAB]
};

// ---- helpers --------------------------------------------------------------

__device__ __forceinline__ v16bf load_b16(const __bf16* p) {
    union { v16bf v; v8bf h[2]; } u;
    u.h[0] = *(const v8bf*)(p);
    u.h[1] = *(const v8bf*)(p + 8);
    return u.v;
}
// A operand: 8 contiguous bf16 at p (K..K+7), 8 at p+16 (K+16..K+23)
__device__ __forceinline__ v16bf load_a16(const __bf16* p) {
    union { v16bf v; v8bf h[2]; } u;
    u.h[0] = *(const v8bf*)(p);
    u.h[1] = *(const v8bf*)(p + 16);
    return u.v;
}
__device__ __forceinline__ v8f wmma_bf16(v16bf a, v16bf b, v8f c) {
    return __builtin_amdgcn_wmma_f32_16x16x32_bf16(
        false, a, false, b, (short)0, c, false, false);
}
__device__ __forceinline__ float sigm(float x) { return 1.f / (1.f + __expf(-x)); }

__device__ __forceinline__ void grid_sync(int* cnt, int* gen) {
    __threadfence();
    __syncthreads();
    if (threadIdx.x == 0) {
        int g = atomicAdd(gen, 0);
        if (atomicAdd(cnt, 1) == NWG - 1) {
            atomicExch(cnt, 0);
            __threadfence();
            atomicAdd(gen, 1);
        } else {
            while (atomicAdd(gen, 0) == g) __builtin_amdgcn_s_sleep(2);
        }
    }
    __syncthreads();
    __threadfence();
}

// 4-gate WMMA step: one A tile against the i/f/g/o B tiles at stride `gs`
#define GATE4(APTR, BPTR, GS)                                   \
    do {                                                        \
        v16bf a_ = load_a16(APTR);                              \
        const __bf16* bp_ = (BPTR);                             \
        a0 = wmma_bf16(a_, load_b16(bp_),            a0);       \
        a1 = wmma_bf16(a_, load_b16(bp_ + (GS)),     a1);       \
        a2 = wmma_bf16(a_, load_b16(bp_ + 2*(GS)),   a2);       \
        a3 = wmma_bf16(a_, load_b16(bp_ + 3*(GS)),   a3);       \
    } while (0)

// ---- prologue kernels -----------------------------------------------------

// Pack W = [wa | wb] (concat along k) into WMMA B layout:
// flat f = ((nt*KT + kt)*32 + lane)*16 + e ;  n = nt*16 + (lane&15),
// k = kt*32 + 8*(lane>>4) + (e&7) + 16*(e>>3).
__global__ void k_pack(const float* wa, const float* wb, int ka, int kb,
                       int KT, int nvalid, int total, __bf16* dst) {
    int f = blockIdx.x * blockDim.x + threadIdx.x;
    if (f >= total) return;
    int e = f & 15;
    int l = (f >> 4) & 31;
    int rest = f >> 9;
    int kt = rest % KT;
    int nt = rest / KT;
    int n = nt * 16 + (l & 15);
    int k = kt * 32 + ((l >> 4) << 3) + (e & 7) + ((e >> 3) << 4);
    float v = 0.f;
    if (n < nvalid) v = (k < ka) ? wa[(size_t)n * ka + k]
                                 : wb[(size_t)n * kb + (k - ka)];
    dst[f] = (__bf16)v;
}

__global__ void k_cvt(const float* src, __bf16* dst, int n) {
    int i = blockIdx.x * blockDim.x + threadIdx.x;
    if (i < n) dst[i] = (__bf16)src[i];
}

__global__ void k_zero(int* p, int n) {
    int i = blockIdx.x * blockDim.x + threadIdx.x;
    if (i < n) p[i] = 0;
}

// ---- main persistent kernel ----------------------------------------------

__global__ __launch_bounds__(NTHR, 1) void k_decoder(DecArgs A) {
    __shared__ float sE[4 * 512];   // energy / attention weights (4 rows)
    __shared__ float sH[4 * 128];   // h2 rows (fp32)

    const int tid  = threadIdx.x;
    const int lane = tid & 31;
    const int wv   = tid >> 5;      // wave in block, 0..7
    const int wg   = blockIdx.x;    // 0..31
    const int hs   = lane >> 4;     // half-select for A layout
    const int m16  = lane & 15;

    for (int t = 0; t < TSTEP; ++t) {
        const int par = t & 1, prv = par ^ 1;
        const __bf16* h1p = A.h1bf + prv * H1N;
        __bf16*       h1c = A.h1bf + par * H1N;
        const __bf16* h2p = A.h2bf + prv * H2N;
        __bf16*       h2c = A.h2bf + par * H2N;
        const __bf16* cxp = A.ctxbf + prv * H2N;
        __bf16*       cxc = A.ctxbf + par * H2N;

        // ---------------- Phase 1: fused LSTM1 GEMM + gates ---------------
        // gates1[B,2048] = [emb | ctx | h1] (B x 1152) @ W1^T, tile (bt,jt).
        // WG owns one col tile jt -> its 8 waves share B tiles via WGP$.
        {
            const int jt = wg;       // output col tile (0..31)
            const int bt = wv;       // batch tile (0..7)
            const int arow = bt * 16 + m16;
            const int eidx = (t == 0) ? 0 : A.text[t * BATCH + arow];
            const __bf16* ebase = A.etab + (size_t)eidx * HDIM + 8 * hs;
            const __bf16* cbase = cxp + arow * VDIM + 8 * hs;
            const __bf16* hbase = h1p + arow * HDIM + 8 * hs;
            const __bf16* bbase = A.w1 + ((size_t)(jt * 36) * 32 + lane) * 16;
            v8f a0 = {}, a1 = {}, a2 = {}, a3 = {};
#pragma unroll 4
            for (int kt = 0; kt < 16; ++kt)                     // k 0..511: embed
                GATE4(ebase + kt * 32, bbase + (size_t)kt * 512, W1G);
#pragma unroll
            for (int kt = 0; kt < 4; ++kt)                      // k 512..639: ctx
                GATE4(cbase + kt * 32, bbase + (size_t)(16 + kt) * 512, W1G);
#pragma unroll 4
            for (int kt = 0; kt < 16; ++kt)                     // k 640..1151: h1
                GATE4(hbase + kt * 32, bbase + (size_t)(20 + kt) * 512, W1G);

            const int col = jt * 16 + m16;
            const float bi = A.b_ih1[col]        + A.b_hh1[col];
            const float bfg= A.b_ih1[512 + col]  + A.b_hh1[512 + col];
            const float bg = A.b_ih1[1024 + col] + A.b_hh1[1024 + col];
            const float bo = A.b_ih1[1536 + col] + A.b_hh1[1536 + col];
#pragma unroll
            for (int v = 0; v < 8; ++v) {
                const int r = bt * 16 + hs * 8 + v;
                const float ig = sigm(a0[v] + bi);
                const float fg = sigm(a1[v] + bfg);
                const float gg = tanhf(a2[v] + bg);
                const float og = sigm(a3[v] + bo);
                const float cn = fg * A.c1[r * HDIM + col] + ig * gg;
                A.c1[r * HDIM + col] = cn;
                h1c[r * HDIM + col] = (__bf16)(og * tanhf(cn));
            }
        }
        grid_sync(A.bar, A.bar + 1);

        // ---------------- Phase 2: fused LSTM2 GEMM + gates ---------------
        // gates2[B,512] = [h1 | h2_prev] (B x 640) @ W2^T
        {
            const int gw = wg * 8 + wv;
            if (gw < 64) {
                const int jt = gw & 7;   // col tile (0..7)
                const int bt = gw >> 3;  // batch tile (0..7)
                const int arow = bt * 16 + m16;
                const __bf16* hbase = h1c + arow * HDIM + 8 * hs;
                const __bf16* pbase = h2p + arow * KDIM + 8 * hs;
                const __bf16* bbase = A.w2 + ((size_t)(jt * 20) * 32 + lane) * 16;
                v8f a0 = {}, a1 = {}, a2 = {}, a3 = {};
#pragma unroll 4
                for (int kt = 0; kt < 16; ++kt)                 // k 0..511: h1
                    GATE4(hbase + kt * 32, bbase + (size_t)kt * 512, W2G);
#pragma unroll
                for (int kt = 0; kt < 4; ++kt)                  // k 512..639: h2
                    GATE4(pbase + kt * 32, bbase + (size_t)(16 + kt) * 512, W2G);

                const int col = jt * 16 + m16;
                const float bi = A.b_ih2[col]       + A.b_hh2[col];
                const float bfg= A.b_ih2[128 + col] + A.b_hh2[128 + col];
                const float bg = A.b_ih2[256 + col] + A.b_hh2[256 + col];
                const float bo = A.b_ih2[384 + col] + A.b_hh2[384 + col];
#pragma unroll
                for (int v = 0; v < 8; ++v) {
                    const int r = bt * 16 + hs * 8 + v;
                    const float ig = sigm(a0[v] + bi);
                    const float fg = sigm(a1[v] + bfg);
                    const float gg = tanhf(a2[v] + bg);
                    const float og = sigm(a3[v] + bo);
                    const float cn = fg * A.c2[r * KDIM + col] + ig * gg;
                    A.c2[r * KDIM + col] = cn;
                    h2c[r * KDIM + col] = (__bf16)(og * tanhf(cn));
                }
            }
        }
        grid_sync(A.bar, A.bar + 1);

        // ---------------- Phase 3: attention (energy, softmax, context) ---
        // Batched per-row matvecs (bsk,bk->bs / bs,bsv->bv): fp32 VALU + LDS.
        {
            const int b0 = wg * 4;
            for (int i = tid; i < 4 * 128; i += NTHR)
                sH[i] = (float)h2c[(b0 + (i >> 7)) * KDIM + (i & 127)];
            __syncthreads();
            for (int i = tid; i < 4 * 512; i += NTHR) {
                const int r = i >> 9, s = i & 511;
                const int b = b0 + r;
                float e = NEGE;
                if (s < SLEN && s < A.lens[b]) {
                    const float* kp = A.key + ((size_t)s * BATCH + b) * KDIM;
                    float acc = 0.f;
#pragma unroll 8
                    for (int k = 0; k < KDIM; ++k) acc += kp[k] * sH[r * 128 + k];
                    e = acc;
                }
                sE[i] = e;
            }
            __syncthreads();
            if (wv < 4) {                     // one wave per batch row
                const int r = wv;
                float m = -3.4e38f;
                for (int s = lane; s < 512; s += 32) m = fmaxf(m, sE[r * 512 + s]);
#pragma unroll
                for (int d = 16; d > 0; d >>= 1) m = fmaxf(m, __shfl_xor(m, d, 32));
                float sum = 0.f;
                for (int s = lane; s < 512; s += 32) {
                    float e = __expf(sE[r * 512 + s] - m);
                    sE[r * 512 + s] = e;
                    sum += e;
                }
#pragma unroll
                for (int d = 16; d > 0; d >>= 1) sum += __shfl_xor(sum, d, 32);
                const float inv = 1.f / sum;
                for (int s = lane; s < 512; s += 32) sE[r * 512 + s] *= inv;
            }
            __syncthreads();
            for (int i = tid; i < 4 * 128; i += NTHR) {
                const int r = i >> 7, vc = i & 127;
                const int b = b0 + r;
                const float* vp = A.values + (size_t)b * VDIM + vc;
                float acc = 0.f;
                for (int s = 0; s < SLEN; ++s)
                    acc += sE[r * 512 + s] * vp[(size_t)s * BATCH * VDIM];
                cxc[b * VDIM + vc] = (__bf16)acc;
            }
        }
        grid_sync(A.bar, A.bar + 1);

        // ---------------- Phase 4: output projection GEMM -----------------
        // pred[B,1000] = [h2 | ctx] (B x 256) @ w_out^T + b_out
        for (int tile = wg * 8 + wv; tile < 8 * 63; tile += NWG * 8) {
            const int nt = tile % 63, bt = tile / 63;
            const int arow = bt * 16 + m16;
            const __bf16* hbase = h2c + arow * KDIM + 8 * hs;
            const __bf16* cbase = cxc + arow * VDIM + 8 * hs;
            const __bf16* bbase = A.w3 + ((size_t)(nt * 8) * 32 + lane) * 16;
            v8f acc = {};
#pragma unroll
            for (int kt = 0; kt < 4; ++kt)                      // k 0..127: h2
                acc = wmma_bf16(load_a16(hbase + kt * 32),
                                load_b16(bbase + (size_t)kt * 512), acc);
#pragma unroll
            for (int kt = 0; kt < 4; ++kt)                      // k 128..255: ctx
                acc = wmma_bf16(load_a16(cbase + kt * 32),
                                load_b16(bbase + (size_t)(4 + kt) * 512), acc);
            const int col = nt * 16 + m16;
            if (col < VOCAB) {
                const float bb = A.b_out[col];
#pragma unroll
                for (int v = 0; v < 8; ++v) {
                    const int r = bt * 16 + hs * 8 + v;
                    A.out[((size_t)r * TSTEP + t) * VOCAB + col] = acc[v] + bb;
                }
            }
        }
        // No barrier here: phase-4 reads (h2c/cxc/w3) are disjoint from the
        // next iteration's phase-1 writes (h1bf[par^1]/c1); all other
        // cross-phase hazards are covered by the 3 barriers above.
    }
}

// ---------------------------------------------------------------------------

extern "C" void kernel_launch(void* const* d_in, const int* in_sizes, int n_in,
                              void* d_out, int out_size, void* d_ws, size_t ws_size,
                              hipStream_t stream) {
    (void)in_sizes; (void)n_in; (void)out_size; (void)ws_size;
    const float* key    = (const float*)d_in[0];
    const float* values = (const float*)d_in[1];
    const int*   lens   = (const int*)d_in[2];
    const int*   text   = (const int*)d_in[3];
    const float* emb    = (const float*)d_in[4];
    const float* w_ih1  = (const float*)d_in[5];
    const float* w_hh1  = (const float*)d_in[6];
    const float* b_ih1  = (const float*)d_in[7];
    const float* b_hh1  = (const float*)d_in[8];
    const float* w_ih2  = (const float*)d_in[9];
    const float* w_hh2  = (const float*)d_in[10];
    const float* b_ih2  = (const float*)d_in[11];
    const float* b_hh2  = (const float*)d_in[12];
    const float* w_out  = (const float*)d_in[13];
    const float* b_out  = (const float*)d_in[14];

    char* ws = (char*)d_ws;
    size_t off = 0;
    auto take = [&](size_t bytes) {
        size_t o = off;
        off = (off + bytes + 255) & ~(size_t)255;
        return o;
    };
    // -- state region (zeroed every call) --
    size_t o_h1  = take((size_t)2 * H1N * 2);
    size_t o_h2  = take((size_t)2 * H2N * 2);
    size_t o_cx  = take((size_t)2 * H2N * 2);
    size_t o_c1  = take((size_t)H1N * 4);
    size_t o_c2  = take((size_t)H2N * 4);
    size_t o_bar = take(256);
    size_t stateEnd = off;
    // -- weights (recomputed every call; deterministic) --
    size_t o_w1 = take((size_t)128 * 36 * 512 * 2);
    size_t o_w2 = take((size_t)32 * 20 * 512 * 2);
    size_t o_w3 = take((size_t)63 * 8 * 512 * 2);
    size_t o_et = take((size_t)VOCAB * HDIM * 2);

    // zero state + barrier
    {
        int n = (int)(stateEnd / 4);
        k_zero<<<(n + 255) / 256, 256, 0, stream>>>((int*)ws, n);
    }
    // pack weights into WMMA B layout (bf16)
    {
        int t1 = 128 * 36 * 512;
        k_pack<<<(t1 + 255) / 256, 256, 0, stream>>>(
            w_ih1, w_hh1, 640, 512, 36, 2048, t1, (__bf16*)(ws + o_w1));
        int t2 = 32 * 20 * 512;
        k_pack<<<(t2 + 255) / 256, 256, 0, stream>>>(
            w_ih2, w_hh2, 512, 128, 20, 512, t2, (__bf16*)(ws + o_w2));
        int t3 = 63 * 8 * 512;
        k_pack<<<(t3 + 255) / 256, 256, 0, stream>>>(
            w_out, w_out, 256, 1, 8, 1000, t3, (__bf16*)(ws + o_w3));
        int te = VOCAB * HDIM;
        k_cvt<<<(te + 255) / 256, 256, 0, stream>>>(emb, (__bf16*)(ws + o_et), te);
    }

    DecArgs A;
    A.key = key; A.values = values; A.lens = lens; A.text = text;
    A.b_ih1 = b_ih1; A.b_hh1 = b_hh1; A.b_ih2 = b_ih2; A.b_hh2 = b_hh2;
    A.b_out = b_out;
    A.w1 = (const __bf16*)(ws + o_w1);
    A.w2 = (const __bf16*)(ws + o_w2);
    A.w3 = (const __bf16*)(ws + o_w3);
    A.etab = (const __bf16*)(ws + o_et);
    A.h1bf = (__bf16*)(ws + o_h1);
    A.h2bf = (__bf16*)(ws + o_h2);
    A.ctxbf = (__bf16*)(ws + o_cx);
    A.c1 = (float*)(ws + o_c1);
    A.c2 = (float*)(ws + o_c2);
    A.bar = (int*)(ws + o_bar);
    A.out = (float*)d_out;

    k_decoder<<<NWG, NTHR, 0, stream>>>(A);
}
